// MultiHeadDepthwiseSelfAttention_4071628996648
// MI455X (gfx1250) — matrix operand order repaired and checked by reading, not compiled, not run on hardware
//
#include <hip/hip_runtime.h>
#include <hip/hip_bf16.h>

#define FEATS 768
#define NSEQ  512
#define BATCH 16
#define HEADS 12
#define DHEAD 64

typedef __attribute__((ext_vector_type(16))) __bf16 v16bf;
typedef __attribute__((ext_vector_type(8)))  __bf16 v8bf;
typedef __attribute__((ext_vector_type(8)))  float  v8f;
typedef __attribute__((ext_vector_type(4)))  float  v4f;

// ---------------------------------------------------------------------------
// WMMA operand loaders (CDNA5 16x16x32 bf16 layouts, ISA 7.12.2)
// A (16x32): lane L holds row L%16; VGPRs0-3 = K {0..7 | 8..15} by lane half,
//            VGPRs4-7 = K {16..23 | 24..31}. Two b128 loads from row-major src.
// ---------------------------------------------------------------------------
__device__ __forceinline__ v16bf load_mat_a(const __bf16* base, int row0, int col0,
                                            int ld, int lane) {
  const int r  = lane & 15;
  const int hi = lane >> 4;
  const __bf16* p = base + (size_t)(row0 + r) * ld + col0 + hi * 8;
  v8bf lo = *reinterpret_cast<const v8bf*>(p);
  v8bf hp = *reinterpret_cast<const v8bf*>(p + 16);
  v16bf out;
#pragma unroll
  for (int i = 0; i < 8; ++i) { out[i] = lo[i]; out[i + 8] = hp[i]; }
  return out;
}

// B (32x16): lane L holds column L%16; K {0..7 | 8..15} then {16..23 | 24..31}
// by lane half. Source must be column-contiguous: element (k,n) at base[n*ldn+k].
__device__ __forceinline__ v16bf load_mat_b(const __bf16* base, int k0, int n0,
                                            int ldn, int lane) {
  const int n  = lane & 15;
  const int hi = lane >> 4;
  const __bf16* p = base + (size_t)(n0 + n) * ldn + k0 + hi * 8;
  v8bf lo = *reinterpret_cast<const v8bf*>(p);
  v8bf hp = *reinterpret_cast<const v8bf*>(p + 16);
  v16bf out;
#pragma unroll
  for (int i = 0; i < 8; ++i) { out[i] = lo[i]; out[i + 8] = hp[i]; }
  return out;
}

__device__ __forceinline__ v8f wmma_bf16(v16bf a, v16bf b, v8f c) {
  return __builtin_amdgcn_wmma_f32_16x16x32_bf16(false, a, false, b,
                                                 (short)0, c, false, false);
}

// ---------------------------------------------------------------------------
// Kernel 1: wo (F,F) fp32 -> bf16
// ---------------------------------------------------------------------------
__global__ void cvt_wo_kernel(const float* __restrict__ wo, __bf16* __restrict__ wob) {
  int i = blockIdx.x * 256 + threadIdx.x;
  wob[i] = (__bf16)wo[i];
}

// ---------------------------------------------------------------------------
// Kernel 2: depthwise conv (K=3, same padding) -> q,k (B,H,N,D) and v^T (B,H,D,N)
// one thread per (b,n,c). The attention scale 1/sqrt(F) is folded into q here.
// ---------------------------------------------------------------------------
__global__ void dwconv_kernel(const float* __restrict__ x,
                              const float* __restrict__ wq, const float* __restrict__ bq,
                              const float* __restrict__ wk, const float* __restrict__ bk,
                              const float* __restrict__ wv, const float* __restrict__ bv,
                              __bf16* __restrict__ q, __bf16* __restrict__ k,
                              __bf16* __restrict__ vt) {
  size_t idx = (size_t)blockIdx.x * 256 + threadIdx.x;   // b*N*F + n*F + c
  int c = (int)(idx % FEATS);
  size_t bn = idx / FEATS;
  int n = (int)(bn % NSEQ);
  int b = (int)(bn / NSEQ);

  float xm = (n > 0)        ? x[idx - FEATS] : 0.f;
  float x0 = x[idx];
  float xp = (n < NSEQ - 1) ? x[idx + FEATS] : 0.f;

  const float scale = 1.0f / sqrtf((float)FEATS);   // reference scales q.k by sqrt(F)
  float qv = (xm * wq[c * 3 + 0] + x0 * wq[c * 3 + 1] + xp * wq[c * 3 + 2] + bq[c]) * scale;
  float kv =  xm * wk[c * 3 + 0] + x0 * wk[c * 3 + 1] + xp * wk[c * 3 + 2] + bk[c];
  float vv =  xm * wv[c * 3 + 0] + x0 * wv[c * 3 + 1] + xp * wv[c * 3 + 2] + bv[c];

  int h = c / DHEAD, d = c % DHEAD;
  size_t bh = (size_t)b * HEADS + h;
  q [(bh * NSEQ + n) * DHEAD + d] = (__bf16)qv;
  k [(bh * NSEQ + n) * DHEAD + d] = (__bf16)kv;
  vt[(bh * DHEAD + d) * NSEQ + n] = (__bf16)vv;
}

// ---------------------------------------------------------------------------
// Kernel 3: flash attention. grid (B*H, N/128), block 256 (8 waves).
// Each wave owns 16 query rows; KV blocks of 64 keys (16 WMMAs per block).
// Softmax row reductions go through an LDS transpose: lane <-> row, so each
// lane reduces 32 contiguous floats locally; only one shfl_xor(16) for max
// and one for sum per block (instead of 64 ds_bpermutes).
// ---------------------------------------------------------------------------
__global__ void flash_attn_kernel(const __bf16* __restrict__ q,
                                  const __bf16* __restrict__ k,
                                  const __bf16* __restrict__ vt,
                                  __bf16* __restrict__ attn) {
  __shared__ float  sS[8][16 * 64];   // per-wave S staging (fp32, C->row transpose)
  __shared__ __bf16 pS[8][16 * 64];   // per-wave P in row-major (A-layout source)
  __shared__ float  aBuf[8][16];      // per-row alpha / inv_l broadcast

  const int bh   = blockIdx.x;
  const int b    = bh / HEADS;
  const int h    = bh % HEADS;
  const int wave = threadIdx.x >> 5;
  const int lane = threadIdx.x & 31;
  const int qrow0 = blockIdx.y * 128 + wave * 16;

  const __bf16* qp = q  + (size_t)bh * NSEQ * DHEAD;
  const __bf16* kp = k  + (size_t)bh * NSEQ * DHEAD;
  const __bf16* vp = vt + (size_t)bh * DHEAD * NSEQ;

  // Q tile (16 x 64, pre-scaled) as two A operands (d-chunks of 32)
  const v16bf aq0 = load_mat_a(qp, qrow0, 0,  DHEAD, lane);
  const v16bf aq1 = load_mat_a(qp, qrow0, 32, DHEAD, lane);

  v8f acc[4] = {};                 // 16 x 64 output accumulator (4 d-tiles)
  const int rowbase = (lane < 16) ? 0 : 8;   // C-layout row base for this lane half
  const int col     = lane & 15;             // C-layout column
  const int rl      = lane & 15;             // softmax-phase row owned by this lane
  const int hl      = lane >> 4;             // which 32-wide half of that row

  float*  sw = sS[wave];
  __bf16* pw = pS[wave];
  float*  aw = aBuf[wave];

  float m_row = -1e30f;            // running stats for row rl (dup in lanes rl,rl+16)
  float l_row = 0.f;

  for (int kb = 0; kb < NSEQ; kb += 64) {
    if (kb + 64 < NSEQ) {          // hint next KV block toward the caches
      __builtin_prefetch(kp + (size_t)(kb + 64) * DHEAD, 0, 3);
      __builtin_prefetch(vp + (size_t)(kb + 64), 0, 3);
    }

    // ---- S = Q K^T (16 x 64) : four 16x16 tiles, K-dim = d (two chunks) ----
    v8f s[4];
#pragma unroll
    for (int t = 0; t < 4; ++t) {
      v16bf b0 = load_mat_b(kp, 0,  kb + t * 16, DHEAD, lane);
      v16bf b1 = load_mat_b(kp, 32, kb + t * 16, DHEAD, lane);
      v8f c = {};
      c = wmma_bf16(aq0, b0, c);
      c = wmma_bf16(aq1, b1, c);
      s[t] = c;
    }

    // ---- scatter S (C-layout) to LDS row-major ----
#pragma unroll
    for (int r = 0; r < 8; ++r)
#pragma unroll
      for (int t = 0; t < 4; ++t)
        sw[(rowbase + r) * 64 + t * 16 + col] = s[t][r];
    asm volatile("s_wait_dscnt 0x0" ::: "memory");

    // ---- per-lane row softmax over 32 contiguous values ----
    float vals[32];
    {
      const v4f* p4 = reinterpret_cast<const v4f*>(sw + rl * 64 + hl * 32);
#pragma unroll
      for (int i = 0; i < 8; ++i) {
        v4f t = p4[i];
        vals[4 * i + 0] = t[0]; vals[4 * i + 1] = t[1];
        vals[4 * i + 2] = t[2]; vals[4 * i + 3] = t[3];
      }
    }
    float mx = vals[0];
#pragma unroll
    for (int i = 1; i < 32; ++i) mx = fmaxf(mx, vals[i]);
    mx = fmaxf(mx, __shfl_xor(mx, 16, 32));          // combine the two row halves

    float mnew  = fmaxf(m_row, mx);
    float alpha = __expf(m_row - mnew);
    m_row = mnew;

    __bf16* prow = pw + rl * 64 + hl * 32;
    float sum = 0.f;
#pragma unroll
    for (int i = 0; i < 4; ++i) {
      v8bf pv;
#pragma unroll
      for (int j = 0; j < 8; ++j) {
        float p = __expf(vals[i * 8 + j] - mnew);
        sum += p;
        pv[j] = (__bf16)p;
      }
      reinterpret_cast<v8bf*>(prow)[i] = pv;
    }
    sum += __shfl_xor(sum, 16, 32);
    l_row = l_row * alpha + sum;

    if (hl == 0) aw[rl] = alpha;                     // publish per-row alpha
    asm volatile("s_wait_dscnt 0x0" ::: "memory");

    // ---- rescale accumulator (C-layout rows rowbase..rowbase+7) ----
    {
      const v4f* ap = reinterpret_cast<const v4f*>(aw + rowbase);
      v4f a0 = ap[0], a1 = ap[1];
      float al[8] = {a0[0], a0[1], a0[2], a0[3], a1[0], a1[1], a1[2], a1[3]};
#pragma unroll
      for (int dt = 0; dt < 4; ++dt)
#pragma unroll
        for (int r = 0; r < 8; ++r) acc[dt][r] *= al[r];
    }

    // ---- O += P V : P from LDS (A-layout), V^T gives contiguous B columns ----
    v16bf pa0 = load_mat_a(pw, 0, 0,  64, lane);
    v16bf pa1 = load_mat_a(pw, 0, 32, 64, lane);
#pragma unroll
    for (int dt = 0; dt < 4; ++dt) {
      v16bf bv0 = load_mat_b(vp, kb,      dt * 16, NSEQ, lane);
      v16bf bv1 = load_mat_b(vp, kb + 32, dt * 16, NSEQ, lane);
      acc[dt] = wmma_bf16(pa0, bv0, acc[dt]);
      acc[dt] = wmma_bf16(pa1, bv1, acc[dt]);
    }
  }

  // ---- publish 1/l per row, normalize, scatter to (B, N, F) bf16 ----
  if (hl == 0) aw[rl] = 1.0f / l_row;
  asm volatile("s_wait_dscnt 0x0" ::: "memory");
  float il[8];
  {
    const v4f* ap = reinterpret_cast<const v4f*>(aw + rowbase);
    v4f a0 = ap[0], a1 = ap[1];
    il[0]=a0[0]; il[1]=a0[1]; il[2]=a0[2]; il[3]=a0[3];
    il[4]=a1[0]; il[5]=a1[1]; il[6]=a1[2]; il[7]=a1[3];
  }
#pragma unroll
  for (int dt = 0; dt < 4; ++dt)
#pragma unroll
    for (int r = 0; r < 8; ++r) {
      size_t n = (size_t)qrow0 + r + rowbase;
      attn[((size_t)b * NSEQ + n) * FEATS + h * DHEAD + dt * 16 + col] =
          (__bf16)(acc[dt][r] * il[r]);
    }
}

// ---------------------------------------------------------------------------
// Kernel 4: out = attn @ wo^T + bo. grid (R/128, F/64), block 256 (8 waves).
// wo row-major (j,k) means B=wo^T has contiguous columns -> load_mat_b directly.
// ---------------------------------------------------------------------------
__global__ void proj_kernel(const __bf16* __restrict__ attn,
                            const __bf16* __restrict__ wob,
                            const float* __restrict__ bo,
                            float* __restrict__ out) {
  const int wave = threadIdx.x >> 5;
  const int lane = threadIdx.x & 31;
  const int row0 = blockIdx.x * 128 + wave * 16;
  const int col0 = blockIdx.y * 64;

  v8f acc[4] = {};
  for (int k0 = 0; k0 < FEATS; k0 += 32) {
    v16bf a = load_mat_a(attn, row0, k0, FEATS, lane);
#pragma unroll
    for (int t = 0; t < 4; ++t) {
      v16bf bb = load_mat_b(wob, k0, col0 + t * 16, FEATS, lane);
      acc[t] = wmma_bf16(a, bb, acc[t]);
    }
  }

  const int rowbase = (lane < 16) ? 0 : 8;
  const int nc      = lane & 15;
#pragma unroll
  for (int t = 0; t < 4; ++t) {
    int c = col0 + t * 16 + nc;
    float bias = bo[c];
#pragma unroll
    for (int r = 0; r < 8; ++r) {
      out[(size_t)(row0 + r + rowbase) * FEATS + c] = acc[t][r] + bias;
    }
  }
}

// ---------------------------------------------------------------------------
extern "C" void kernel_launch(void* const* d_in, const int* in_sizes, int n_in,
                              void* d_out, int out_size, void* d_ws, size_t ws_size,
                              hipStream_t stream) {
  (void)in_sizes; (void)n_in; (void)out_size; (void)ws_size;

  const float* x  = (const float*)d_in[0];
  const float* wq = (const float*)d_in[1];
  const float* bq = (const float*)d_in[2];
  const float* wk = (const float*)d_in[3];
  const float* bk = (const float*)d_in[4];
  const float* wv = (const float*)d_in[5];
  const float* bv = (const float*)d_in[6];
  const float* wo = (const float*)d_in[7];
  const float* bo = (const float*)d_in[8];
  float* out = (float*)d_out;

  // workspace carve-up (all sizes 256B-aligned)
  const size_t qkvElems = (size_t)BATCH * HEADS * NSEQ * DHEAD;  // 6,291,456
  char* ws = (char*)d_ws;
  __bf16* qb   = (__bf16*)(ws);
  __bf16* kb   = (__bf16*)(ws + qkvElems * 2);
  __bf16* vtb  = (__bf16*)(ws + qkvElems * 4);
  __bf16* atnb = (__bf16*)(ws + qkvElems * 6);
  __bf16* wob  = (__bf16*)(ws + qkvElems * 8);

  // 1. wo -> bf16
  cvt_wo_kernel<<<(FEATS * FEATS) / 256, 256, 0, stream>>>(wo, wob);

  // 2. depthwise conv q/k/v (+ head split, V transpose, q pre-scale)
  dwconv_kernel<<<(BATCH * NSEQ * FEATS) / 256, 256, 0, stream>>>(
      x, wq, bq, wk, bk, wv, bv, qb, kb, vtb);

  // 3. flash attention
  flash_attn_kernel<<<dim3(BATCH * HEADS, NSEQ / 128), 256, 0, stream>>>(
      qb, kb, vtb, atnb);

  // 4. output projection
  proj_kernel<<<dim3((BATCH * NSEQ) / 128, FEATS / 64), 256, 0, stream>>>(
      atnb, wob, bo, out);
}
